// MultiHeadAttention_29497835389033
// MI455X (gfx1250) — compile-verified, matrix operand order checked by reading, not run on hardware
//
#include <hip/hip_runtime.h>
#include <hip/hip_bf16.h>

typedef __attribute__((ext_vector_type(16))) __bf16    v16bf;
typedef __attribute__((ext_vector_type(8)))  float     v8f;
typedef __attribute__((ext_vector_type(4)))  unsigned  v4u;

union Frag { v16bf bf; v4u u[2]; };

static __device__ __forceinline__ unsigned short f2bf(float f) {
    union { float f; unsigned u; } c; c.f = f;
    unsigned u = c.u;
    unsigned r = u + 0x7FFFu + ((u >> 16) & 1u);
    return (unsigned short)(r >> 16);
}

static __device__ __forceinline__ v8f wmma_bf16(v16bf a, v16bf b, v8f c) {
    return __builtin_amdgcn_wmma_f32_16x16x32_bf16(
        false, a, false, b, (short)0, c, false, false);
}

// ---------------------------------------------------------------------------
// fp32 -> bf16 conversion (grid-stride)
// ---------------------------------------------------------------------------
__global__ void cvt_f32_bf16(const float* __restrict__ in,
                             unsigned short* __restrict__ out, int n) {
    int i = blockIdx.x * blockDim.x + threadIdx.x;
    int stride = gridDim.x * blockDim.x;
    for (; i < n; i += stride) out[i] = f2bf(in[i]);
}

// ---------------------------------------------------------------------------
// WMMA GEMM: C[4096,1024] = A[4096,1024] * B[1024,1024]^T + bias[1024]
//   Wave tile 32x64 (8 WMMA per K-step), WG tile 64x256 (2x4 waves),
//   grid = 64 (M) * 4 (N) = 256 blocks.  K-loop is 2x unrolled with
//   ping-pong fragment tiles so loads overlap WMMA execution.
//   mode 0: bf16 out, [B,H,S,D]   (Q, K)
//   mode 1: bf16 out, [B,H,D,S]   (V transposed)
//   mode 2: f32  out, [4096,1024] (final projection)
// ---------------------------------------------------------------------------
struct Tile { Frag a[2]; Frag b[4]; };

static __device__ __forceinline__ void load_tile(
    Tile& t,
    const unsigned short* a0, const unsigned short* a1,
    const unsigned short* b0, const unsigned short* b1,
    const unsigned short* b2, const unsigned short* b3,
    int kk, int kh) {
    const int c0 = kk + kh * 8, c1 = kk + 16 + kh * 8;
    t.a[0].u[0] = *(const v4u*)(a0 + c0); t.a[0].u[1] = *(const v4u*)(a0 + c1);
    t.a[1].u[0] = *(const v4u*)(a1 + c0); t.a[1].u[1] = *(const v4u*)(a1 + c1);
    t.b[0].u[0] = *(const v4u*)(b0 + c0); t.b[0].u[1] = *(const v4u*)(b0 + c1);
    t.b[1].u[0] = *(const v4u*)(b1 + c0); t.b[1].u[1] = *(const v4u*)(b1 + c1);
    t.b[2].u[0] = *(const v4u*)(b2 + c0); t.b[2].u[1] = *(const v4u*)(b2 + c1);
    t.b[3].u[0] = *(const v4u*)(b3 + c0); t.b[3].u[1] = *(const v4u*)(b3 + c1);
}

static __device__ __forceinline__ void mma_tile(v8f (&acc)[2][4], const Tile& t) {
#pragma unroll
    for (int i = 0; i < 2; ++i)
#pragma unroll
        for (int j = 0; j < 4; ++j)
            acc[i][j] = wmma_bf16(t.a[i].bf, t.b[j].bf, acc[i][j]);
}

__global__ __launch_bounds__(256) void gemm_bf16(
    const unsigned short* __restrict__ A,
    const unsigned short* __restrict__ B,
    const float* __restrict__ bias,
    unsigned short* __restrict__ obf,
    float* __restrict__ of32,
    int mode) {
    const int l  = threadIdx.x & 31;
    const int w  = threadIdx.x >> 5;
    const int lh = l & 15;
    const int kh = l >> 4;

    const int mblk  = blockIdx.x & 63;
    const int nblk  = blockIdx.x >> 6;
    const int mbase = mblk * 64 + (w & 1) * 32;
    const int nbase = nblk * 256 + (w >> 1) * 64;

    const unsigned short* a0 = A + (size_t)(mbase + lh) * 1024;
    const unsigned short* a1 = A + (size_t)(mbase + 16 + lh) * 1024;
    const unsigned short* b0 = B + (size_t)(nbase + lh) * 1024;
    const unsigned short* b1 = B + (size_t)(nbase + 16 + lh) * 1024;
    const unsigned short* b2 = B + (size_t)(nbase + 32 + lh) * 1024;
    const unsigned short* b3 = B + (size_t)(nbase + 48 + lh) * 1024;

    v8f acc[2][4] = {};
    Tile t0, t1;
    load_tile(t0, a0, a1, b0, b1, b2, b3, 0, kh);

    for (int kk = 0; kk < 1024; kk += 64) {
        load_tile(t1, a0, a1, b0, b1, b2, b3, kk + 32, kh);
        __builtin_prefetch(a0 + kk + 128, 0, 1);
        mma_tile(acc, t0);
        if (kk + 64 < 1024)
            load_tile(t0, a0, a1, b0, b1, b2, b3, kk + 64, kh);
        mma_tile(acc, t1);
    }

#pragma unroll
    for (int i = 0; i < 2; ++i) {
#pragma unroll
        for (int j = 0; j < 4; ++j) {
            const int n  = nbase + j * 16 + lh;
            const float bn = bias[n];
#pragma unroll
            for (int r = 0; r < 8; ++r) {
                const int m = mbase + i * 16 + r + 8 * kh;
                const float v = acc[i][j][r] + bn;
                if (mode == 2) {
                    of32[(size_t)m * 1024 + n] = v;
                } else {
                    const int b = m >> 10, s = m & 1023;
                    const int h = n >> 6,  d = n & 63;
                    size_t idx = (mode == 0)
                        ? (((size_t)(b * 16 + h) * 1024 + s) * 64 + d)   // [B,H,S,D]
                        : (((size_t)(b * 16 + h) * 64 + d) * 1024 + s);  // [B,H,D,S]
                    obf[idx] = f2bf(v);
                }
            }
        }
    }
}

// ---------------------------------------------------------------------------
// Flash attention: Q[B,H,S,D], K[B,H,S,D], Vt[B,H,D,S] (bf16) -> O[4096,1024]
// One wave = 16 q rows; 8 waves/WG; grid = (B*H)*(S/128) = 512 blocks.
// K-tiles ping-pong preloaded one block ahead; V fragments issued at the top
// of each block so softmax VALU + LDS P round-trip hide their latency.
// ---------------------------------------------------------------------------
struct KT { Frag b[4]; };

static __device__ __forceinline__ void load_kt(
    KT& t, const unsigned short* Kh, int kb, int lh, int kh) {
    const unsigned short* kr0 = Kh + (size_t)(kb + lh) * 64;
    const unsigned short* kr1 = Kh + (size_t)(kb + 16 + lh) * 64;
    t.b[0].u[0] = *(const v4u*)(kr0 + kh * 8);
    t.b[0].u[1] = *(const v4u*)(kr0 + 16 + kh * 8);
    t.b[1].u[0] = *(const v4u*)(kr0 + 32 + kh * 8);
    t.b[1].u[1] = *(const v4u*)(kr0 + 48 + kh * 8);
    t.b[2].u[0] = *(const v4u*)(kr1 + kh * 8);
    t.b[2].u[1] = *(const v4u*)(kr1 + 16 + kh * 8);
    t.b[3].u[0] = *(const v4u*)(kr1 + 32 + kh * 8);
    t.b[3].u[1] = *(const v4u*)(kr1 + 48 + kh * 8);
}

__global__ __launch_bounds__(256) void attn_flash(
    const unsigned short* __restrict__ Q,
    const unsigned short* __restrict__ K,
    const unsigned short* __restrict__ Vt,
    unsigned short* __restrict__ O) {
    __shared__ __align__(16) unsigned short pbuf[8 * 16 * 32];

    const int l  = threadIdx.x & 31;
    const int w  = threadIdx.x >> 5;
    const int lh = l & 15;
    const int kh = l >> 4;

    const int bh    = blockIdx.x >> 3;   // b*16 + h
    const int qblk  = blockIdx.x & 7;
    const int qbase = qblk * 128 + w * 16;

    const unsigned short* Qh  = Q  + (size_t)bh * 1024 * 64;
    const unsigned short* Kh  = K  + (size_t)bh * 1024 * 64;
    const unsigned short* Vth = Vt + (size_t)bh * 64 * 1024;
    unsigned short* pw = pbuf + w * 16 * 32;

    Frag aq[2];
    {
        const unsigned short* qr = Qh + (size_t)(qbase + lh) * 64;
        aq[0].u[0] = *(const v4u*)(qr + kh * 8);
        aq[0].u[1] = *(const v4u*)(qr + 16 + kh * 8);
        aq[1].u[0] = *(const v4u*)(qr + 32 + kh * 8);
        aq[1].u[1] = *(const v4u*)(qr + 48 + kh * 8);
    }

    float mrow[8], lrow[8];
    v8f oacc[4] = {};
#pragma unroll
    for (int r = 0; r < 8; ++r) { mrow[r] = -3.0e38f; lrow[r] = 0.0f; }

    const float scale = 0.125f;  // 1/sqrt(64)

    auto body = [&](int kb, const KT& kt) {
        // V fragments issued early: softmax + LDS round-trip hide the latency
        Frag vf[4];
#pragma unroll
        for (int dt = 0; dt < 4; ++dt) {
            const unsigned short* vr = Vth + (size_t)(dt * 16 + lh) * 1024 + kb;
            vf[dt].u[0] = *(const v4u*)(vr + kh * 8);
            vf[dt].u[1] = *(const v4u*)(vr + 16 + kh * 8);
        }

        // scores: two 16x16 tiles over this 32-wide sk block
        v8f c0 = {}, c1 = {};
        c0 = wmma_bf16(aq[0].bf, kt.b[0].bf, c0);
        c0 = wmma_bf16(aq[1].bf, kt.b[1].bf, c0);
        c1 = wmma_bf16(aq[0].bf, kt.b[2].bf, c1);
        c1 = wmma_bf16(aq[1].bf, kt.b[3].bf, c1);

        // online softmax update
        float bm[8];
#pragma unroll
        for (int r = 0; r < 8; ++r) {
            float s0 = c0[r] * scale, s1 = c1[r] * scale;
            c0[r] = s0; c1[r] = s1;
            bm[r] = fmaxf(s0, s1);
        }
#pragma unroll
        for (int r = 0; r < 8; ++r) {
            bm[r] = fmaxf(bm[r], __shfl_xor(bm[r], 1, 32));
            bm[r] = fmaxf(bm[r], __shfl_xor(bm[r], 2, 32));
            bm[r] = fmaxf(bm[r], __shfl_xor(bm[r], 4, 32));
            bm[r] = fmaxf(bm[r], __shfl_xor(bm[r], 8, 32));
        }
#pragma unroll
        for (int r = 0; r < 8; ++r) {
            const float mn   = fmaxf(mrow[r], bm[r]);
            const float corr = __expf(mrow[r] - mn);
            const float e0   = __expf(c0[r] - mn);
            const float e1   = __expf(c1[r] - mn);
            float ps = e0 + e1;
            ps += __shfl_xor(ps, 1, 32);
            ps += __shfl_xor(ps, 2, 32);
            ps += __shfl_xor(ps, 4, 32);
            ps += __shfl_xor(ps, 8, 32);
            lrow[r] = lrow[r] * corr + ps;
            mrow[r] = mn;
#pragma unroll
            for (int dt = 0; dt < 4; ++dt) oacc[dt][r] *= corr;
            // stash P (bf16) in C-layout: row = r + 8*kh, col = t*16 + lh
            pw[(r + 8 * kh) * 32 + lh]      = f2bf(e0);
            pw[(r + 8 * kh) * 32 + 16 + lh] = f2bf(e1);
        }

        // reload P in A-fragment layout (same-wave DS ops are in-order)
        Frag pa;
        pa.u[0] = *(const v4u*)(pw + lh * 32 + kh * 8);
        pa.u[1] = *(const v4u*)(pw + lh * 32 + 16 + kh * 8);

        // O += P * V
#pragma unroll
        for (int dt = 0; dt < 4; ++dt)
            oacc[dt] = wmma_bf16(pa.bf, vf[dt].bf, oacc[dt]);
    };

    KT k0, k1;
    load_kt(k0, Kh, 0, lh, kh);
    for (int kb = 0; kb < 1024; kb += 64) {
        load_kt(k1, Kh, kb + 32, lh, kh);
        body(kb, k0);
        load_kt(k0, Kh, (kb + 64) & 1023, lh, kh);
        body(kb + 32, k1);
    }

    // epilogue: normalize and write head-concatenated O
    const int b = bh >> 4, h = bh & 15;
#pragma unroll
    for (int r = 0; r < 8; ++r) {
        const float inv = 1.0f / lrow[r];
        const int m = qbase + r + 8 * kh;
#pragma unroll
        for (int dt = 0; dt < 4; ++dt) {
            const int col = h * 64 + dt * 16 + lh;
            O[((size_t)(b * 1024 + m)) * 1024 + col] = f2bf(oacc[dt][r] * inv);
        }
    }
}

// ---------------------------------------------------------------------------
// Host-side orchestration
// ---------------------------------------------------------------------------
extern "C" void kernel_launch(void* const* d_in, const int* in_sizes, int n_in,
                              void* d_out, int out_size, void* d_ws, size_t ws_size,
                              hipStream_t stream) {
    const float* X  = (const float*)d_in[0];
    const float* Wq = (const float*)d_in[1];
    const float* bq = (const float*)d_in[2];
    const float* Wk = (const float*)d_in[3];
    const float* bk = (const float*)d_in[4];
    const float* Wv = (const float*)d_in[5];
    const float* bv = (const float*)d_in[6];
    const float* Wo = (const float*)d_in[7];
    const float* bo = (const float*)d_in[8];
    float* out = (float*)d_out;

    char* ws = (char*)d_ws;
    size_t off = 0;
    auto carve = [&](size_t bytes) {
        char* p = ws + off;
        off = (off + bytes + 255) & ~(size_t)255;
        return p;
    };
    const size_t M = 4096, E = 1024;
    unsigned short* Xb  = (unsigned short*)carve(M * E * 2);
    unsigned short* Wqb = (unsigned short*)carve(E * E * 2);
    unsigned short* Wkb = (unsigned short*)carve(E * E * 2);
    unsigned short* Wvb = (unsigned short*)carve(E * E * 2);
    unsigned short* Wob = (unsigned short*)carve(E * E * 2);
    unsigned short* Qb  = (unsigned short*)carve(M * E * 2);   // [B,H,S,D]
    unsigned short* Kb  = (unsigned short*)carve(M * E * 2);   // [B,H,S,D]
    unsigned short* Vtb = (unsigned short*)carve(M * E * 2);   // [B,H,D,S]
    unsigned short* Ob  = (unsigned short*)carve(M * E * 2);   // [4096,1024]

    cvt_f32_bf16<<<1024, 256, 0, stream>>>(X,  Xb,  (int)(M * E));
    cvt_f32_bf16<<<512,  256, 0, stream>>>(Wq, Wqb, (int)(E * E));
    cvt_f32_bf16<<<512,  256, 0, stream>>>(Wk, Wkb, (int)(E * E));
    cvt_f32_bf16<<<512,  256, 0, stream>>>(Wv, Wvb, (int)(E * E));
    cvt_f32_bf16<<<512,  256, 0, stream>>>(Wo, Wob, (int)(E * E));

    gemm_bf16<<<256, 256, 0, stream>>>(Xb, Wqb, bq, Qb,  nullptr, 0);
    gemm_bf16<<<256, 256, 0, stream>>>(Xb, Wkb, bk, Kb,  nullptr, 0);
    gemm_bf16<<<256, 256, 0, stream>>>(Xb, Wvb, bv, Vtb, nullptr, 1);

    attn_flash<<<512, 256, 0, stream>>>(Qb, Kb, Vtb, Ob);

    gemm_bf16<<<256, 256, 0, stream>>>(Ob, Wob, bo, nullptr, out, 2);
}
